// ResNet_cifar10_38637525794934
// MI455X (gfx1250) — 8-bit matrix operand order repaired and checked by reading, compiled, not run on hardware
//
#include <hip/hip_runtime.h>
#include <stdint.h>


typedef __attribute__((ext_vector_type(8))) int v8i;

static __device__ __forceinline__ int iclamp(int v, int lo, int hi) {
  v = v < lo ? lo : v;
  return v > hi ? hi : v;
}

__global__ void k_binarize(const float* __restrict__ x, char* __restrict__ xb,
                           int N, int Cin, int Cpad, int H, int W) {
  int Hp = H + 2, Wp = W + 2;
  long long i = (long long)blockIdx.x * blockDim.x + threadIdx.x;
  long long total = (long long)N * Cpad * Hp * Wp;
  if (i >= total) return;
  int xw = (int)(i % Wp); long long r = i / Wp;
  int yh = (int)(r % Hp); r /= Hp;
  int c  = (int)(r % Cpad);
  int n  = (int)(r / Cpad);
  char v = 0;
  if (c < Cin && yh >= 1 && yh <= H && xw >= 1 && xw <= W) {
    float f = x[(((long long)n * Cin + c) * H + (yh - 1)) * W + (xw - 1)];
    v = (f >= 0.0f) ? (char)1 : (char)-1;
  }
  xb[i] = v;
}

__global__ void k_repack_w(const float* __restrict__ w, char* __restrict__ wpk,
                           int Cout, int Cin, int taps, int g, int G) {
  long long i = (long long)blockIdx.x * blockDim.x + threadIdx.x;
  long long total = (long long)(Cout / 16) * G * 1024;
  if (i >= total) return;
  int byte = (int)(i & 31);
  int lane = (int)((i >> 5) & 31);
  long long rest = i >> 10;
  int grp = (int)(rest % G);
  int ct  = (int)(rest / G);
  int h = lane >> 4, M = lane & 15;
  int d = byte >> 2, j = byte & 3;
  int k = (d >> 1) * 16 + h * 8 + (d & 1) * 4 + j;
  int c = k / taps, t = k - c * taps;
  int cin = grp * g + c;
  char v = 0;
  if (c < g && cin < Cin) {
    float wv = w[((long long)(ct * 16 + M) * Cin + cin) * taps + t];
    v = (wv >= 0.0f) ? (char)1 : (char)-1;
  }
  wpk[i] = v;
}

template <int CTW>
__global__ __launch_bounds__(32) void k_binconv(
    const char* __restrict__ xb, const char* __restrict__ wpk,
    float* __restrict__ out,
    int Cpad, int Hp, int Wp, int Cout, int Ho, int Wo,
    int stride, int padAdj, int g, int taps, int KW, int G) {
  int lane = threadIdx.x;
  int tileP = blockIdx.x;
  int ct0 = blockIdx.y * CTW;

  int col = lane & 15;
  int h = lane >> 4;
  int HoWo = Ho * Wo;
  int pix = tileP * 16 + col;
  int n = pix / HoWo;
  int rem = pix - n * HoWo;
  int oy = rem / Wo;
  int ox = rem - oy * Wo;
  int HpWp = Hp * Wp;
  long long base = (long long)n * Cpad * HpWp +
                   (long long)(oy * stride + padAdj) * Wp + (ox * stride + padAdj);
  int grpStride = g * HpWp;

  int offs[32];
#pragma unroll
  for (int d = 0; d < 8; ++d) {
#pragma unroll
    for (int j = 0; j < 4; ++j) {
      int k = (d >> 1) * 16 + h * 8 + (d & 1) * 4 + j;
      int c = k / taps, t = k - c * taps;
      offs[d * 4 + j] = c * HpWp + (t / KW) * Wp + (t % KW);
    }
  }

  v8i acc[CTW];
#pragma unroll
  for (int c = 0; c < CTW; ++c) {
#pragma unroll
    for (int e = 0; e < 8; ++e) acc[c][e] = 0;
  }
  v8i czero = {0, 0, 0, 0, 0, 0, 0, 0};

  const char* xptr = xb + base;
  const char* wptr = wpk + (lane << 5) + ((long long)ct0 * G << 10);
  long long tileStride = (long long)G << 10;

  auto do_group = [&](v8i (&pout)[CTW]) {
    v8i bfrag;
#pragma unroll
    for (int d = 0; d < 8; ++d) {
      unsigned b0 = (unsigned char)xptr[offs[d * 4 + 0]];
      unsigned b1 = (unsigned char)xptr[offs[d * 4 + 1]];
      unsigned b2 = (unsigned char)xptr[offs[d * 4 + 2]];
      unsigned b3 = (unsigned char)xptr[offs[d * 4 + 3]];
      bfrag[d] = (int)(b0 | (b1 << 8) | (b2 << 16) | (b3 << 24));
    }
    __builtin_prefetch(xptr + grpStride, 0, 3);
#pragma unroll
    for (int c = 0; c < CTW; ++c) {
      const int4* wp = (const int4*)(wptr + (long long)c * tileStride);
      int4 w0 = wp[0], w1 = wp[1];
      v8i af;
      af[0] = w0.x; af[1] = w0.y; af[2] = w0.z; af[3] = w0.w;
      af[4] = w1.x; af[5] = w1.y; af[6] = w1.z; af[7] = w1.w;
      pout[c] = __builtin_amdgcn_wmma_i32_16x16x64_iu8(true, af, true, bfrag,
                                                       czero, false, false);
    }
    xptr += grpStride;
    wptr += 1024;
  };

  auto do_update = [&](v8i (&p)[CTW]) {
#pragma unroll
    for (int c = 0; c < CTW; ++c) {
#pragma unroll
      for (int e = 0; e < 8; ++e) {
        int a = acc[c][e] + p[c][e];
        a &= ~1;
        acc[c][e] = iclamp(a, -256, 254);
      }
    }
  };

  v8i pA[CTW], pB[CTW];
  do_group(pA);
  int grp = 1;
  for (; grp + 1 < G; grp += 2) {
    do_group(pB);
    do_update(pA);
    do_group(pA);
    do_update(pB);
  }
  if (grp < G) {
    do_group(pB);
    do_update(pA);
    do_update(pB);
  } else {
    do_update(pA);
  }

#pragma unroll
  for (int c = 0; c < CTW; ++c) {
    int coutBase = (ct0 + c) * 16 + h * 8;
#pragma unroll
    for (int e = 0; e < 8; ++e) {
      long long o = (((long long)n * Cout + (coutBase + e)) * Ho + oy) * Wo + ox;
      out[o] = (float)acc[c][e];
    }
  }
}

__global__ void k_conv1(const float* __restrict__ x, const float* __restrict__ w,
                        float* __restrict__ out, int N, int Cout, int H, int W) {
  long long i = (long long)blockIdx.x * blockDim.x + threadIdx.x;
  long long total = (long long)N * Cout * H * W;
  if (i >= total) return;
  int ox = (int)(i % W); long long r = i / W;
  int oy = (int)(r % H); r /= H;
  int co = (int)(r % Cout);
  int n  = (int)(r / Cout);
  float s = 0.0f;
  for (int ci = 0; ci < 3; ++ci)
    for (int ty = 0; ty < 3; ++ty) {
      int iy = oy + ty - 1;
      if (iy < 0 || iy >= H) continue;
      for (int tx = 0; tx < 3; ++tx) {
        int ix = ox + tx - 1;
        if (ix < 0 || ix >= W) continue;
        s += x[(((long long)n * 3 + ci) * H + iy) * W + ix] *
             w[((co * 3 + ci) * 3 + ty) * 3 + tx];
      }
    }
  out[i] = s;
}

__global__ __launch_bounds__(256) void k_bnstats(const float* __restrict__ x,
                                                 float* __restrict__ mean,
                                                 float* __restrict__ var,
                                                 int N, int C, int HW) {
  int c = blockIdx.x;
  int tid = threadIdx.x;
  long long cnt = (long long)N * HW;
  float s = 0.0f, ss = 0.0f;
  for (long long i = tid; i < cnt; i += 256) {
    long long n = i / HW, p = i - n * HW;
    float v = x[(n * C + c) * HW + p];
    s += v; ss += v * v;
  }
  __shared__ float sh[256], sh2[256];
  sh[tid] = s; sh2[tid] = ss;
  __syncthreads();
  for (int o = 128; o > 0; o >>= 1) {
    if (tid < o) { sh[tid] += sh[tid + o]; sh2[tid] += sh2[tid + o]; }
    __syncthreads();
  }
  if (tid == 0) {
    float m = sh[0] / (float)cnt;
    mean[c] = m;
    var[c] = sh2[0] / (float)cnt - m * m;
  }
}

__global__ void k_bnapply(float* __restrict__ x, const float* __restrict__ mean,
                          const float* __restrict__ var, const float* __restrict__ gm,
                          const float* __restrict__ bt, int C, int HW,
                          long long total, int doTanh) {
  long long i = (long long)blockIdx.x * blockDim.x + threadIdx.x;
  if (i >= total) return;
  int c = (int)((i / HW) % C);
  float v = (x[i] - mean[c]) * rsqrtf(var[c] + 1e-5f) * gm[c] + bt[c];
  if (doTanh) v = fminf(1.0f, fmaxf(-1.0f, v));
  x[i] = v;
}

__global__ void k_add(float* __restrict__ d, const float* __restrict__ s,
                      long long total) {
  long long i = (long long)blockIdx.x * blockDim.x + threadIdx.x;
  if (i >= total) return;
  d[i] += s[i];
}

__global__ void k_avgpool(const float* __restrict__ x, float* __restrict__ o,
                          int N, int C, int HW) {
  int i = blockIdx.x * blockDim.x + threadIdx.x;
  if (i >= N * C) return;
  const float* p = x + (long long)i * HW;
  float s = 0.0f;
  for (int j = 0; j < HW; ++j) s += p[j];
  o[i] = s / (float)HW;
}

__global__ void k_fc(const float* __restrict__ x, const float* __restrict__ w,
                     const float* __restrict__ b, float* __restrict__ o,
                     int N, int Cf, int Co) {
  int i = blockIdx.x * blockDim.x + threadIdx.x;
  if (i >= N * Co) return;
  int n = i / Co, j = i - n * Co;
  float s = b[j];
  for (int k = 0; k < Cf; ++k) s += x[(long long)n * Cf + k] * w[(long long)j * Cf + k];
  o[i] = s;
}

__global__ void k_logsoftmax(const float* __restrict__ x, float* __restrict__ o,
                             int N, int Co) {
  int n = blockIdx.x * blockDim.x + threadIdx.x;
  if (n >= N) return;
  float m = -3.4e38f;
  for (int j = 0; j < Co; ++j) m = fmaxf(m, x[n * Co + j]);
  float s = 0.0f;
  for (int j = 0; j < Co; ++j) s += expf(x[n * Co + j] - m);
  float l = logf(s);
  for (int j = 0; j < Co; ++j) o[n * Co + j] = x[n * Co + j] - m - l;
}

extern "C" void kernel_launch(void* const* d_in, const int* in_sizes, int n_in,
                              void* d_out, int out_size, void* d_ws, size_t ws_size,
                              hipStream_t stream) {
  (void)in_sizes; (void)n_in; (void)out_size; (void)ws_size;

  int ii = 0;
  const float* X      = (const float*)d_in[ii++];
  const float* WCONV1 = (const float*)d_in[ii++];
  const float* BN1G   = (const float*)d_in[ii++];
  const float* BN1B   = (const float*)d_in[ii++];
  struct BP { const float *w1,*b1g,*b1b,*w2,*b2g,*b2b,*dsw,*dsg,*dsb; };
  BP bp[6];
  for (int s = 0; s < 3; ++s)
    for (int b = 0; b < 2; ++b) {
      BP& p = bp[s * 2 + b];
      p.w1  = (const float*)d_in[ii++];
      p.b1g = (const float*)d_in[ii++];
      p.b1b = (const float*)d_in[ii++];
      p.w2  = (const float*)d_in[ii++];
      p.b2g = (const float*)d_in[ii++];
      p.b2b = (const float*)d_in[ii++];
      if (s > 0 && b == 0) {
        p.dsw = (const float*)d_in[ii++];
        p.dsg = (const float*)d_in[ii++];
        p.dsb = (const float*)d_in[ii++];
      } else {
        p.dsw = p.dsg = p.dsb = nullptr;
      }
    }
  const float* BN2G = (const float*)d_in[ii++];
  const float* BN2B = (const float*)d_in[ii++];
  const float* FCW  = (const float*)d_in[ii++];
  const float* FCB  = (const float*)d_in[ii++];
  const float* BN3G = (const float*)d_in[ii++];
  const float* BN3B = (const float*)d_in[ii++];

  const int N = 256;
  char* ws = (char*)d_ws;
  size_t off = 0;
  auto carve = [&](size_t bytes) -> char* {
    char* p = ws + off;
    off += (bytes + 255) & ~(size_t)255;
    return p;
  };
  const size_t SZ_F = (size_t)N * 80 * 32 * 32 * 4;
  float* F0   = (float*)carve(SZ_F);
  float* F1   = (float*)carve(SZ_F);
  float* F2   = (float*)carve((size_t)N * 160 * 16 * 16 * 4);
  char*  XB   = carve((size_t)N * 128 * 34 * 34 + (1 << 20));
  char*  WPK  = carve(1 << 20);
  float* MEAN = (float*)carve(4096);
  float* VAR  = (float*)carve(4096);
  float* POOL = (float*)carve((size_t)N * 320 * 4);
  float* FCO  = (float*)carve((size_t)N * 10 * 4);

  auto cdiv = [](int a, int b) { return (a + b - 1) / b; };

  auto binarize = [&](const float* src, int Cin, int Cpad, int H) {
    long long total = (long long)N * Cpad * (H + 2) * (H + 2);
    k_binarize<<<(unsigned)((total + 255) / 256), 256, 0, stream>>>(
        src, XB, N, Cin, Cpad, H, H);
  };

  auto binconv = [&](int Cpad, const float* wf, float* out, int Cin, int H,
                     int Cout, int K, int stride, int pad) {
    int taps = K * K, g = 64 / taps, G = cdiv(Cin, g);
    int Ho = (H + 2 * pad - K) / stride + 1;
    int ctTiles = Cout / 16;
    long long rp = (long long)ctTiles * G * 1024;
    k_repack_w<<<(unsigned)((rp + 255) / 256), 256, 0, stream>>>(
        wf, WPK, Cout, Cin, taps, g, G);
    dim3 grid((unsigned)(N * Ho * Ho / 16), (unsigned)(ctTiles / 5));
    k_binconv<5><<<grid, 32, 0, stream>>>(XB, WPK, out, Cpad, H + 2, H + 2,
                                          Cout, Ho, Ho, stride, 1 - pad, g,
                                          taps, K, G);
  };

  auto bnorm = [&](float* x, const float* gm, const float* bt, int C, int HW,
                   int doTanh) {
    k_bnstats<<<(unsigned)C, 256, 0, stream>>>(x, MEAN, VAR, N, C, HW);
    long long total = (long long)N * C * HW;
    k_bnapply<<<(unsigned)((total + 255) / 256), 256, 0, stream>>>(
        x, MEAN, VAR, gm, bt, C, HW, total, doTanh);
  };

  float* cur = F0;
  float* nxt = F1;
  {
    long long total = (long long)N * 80 * 32 * 32;
    k_conv1<<<(unsigned)((total + 255) / 256), 256, 0, stream>>>(
        X, WCONV1, cur, N, 80, 32, 32);
    bnorm(cur, BN1G, BN1B, 80, 32 * 32, 1);
  }

  const int cfg[6][6] = {
      {80, 80, 32, 1, 0, 1},  {80, 80, 32, 1, 0, 1},
      {80, 160, 32, 2, 1, 1}, {160, 160, 16, 1, 0, 1},
      {160, 320, 16, 2, 1, 1},{320, 320, 8, 1, 0, 0},
  };
  for (int bi = 0; bi < 6; ++bi) {
    int Cin = cfg[bi][0], Cout = cfg[bi][1], H = cfg[bi][2];
    int st = cfg[bi][3], hasDs = cfg[bi][4], dbt = cfg[bi][5];
    int Ho = (H - 1) / st + 1;
    const BP& p = bp[bi];

    int Cpad = 7 * cdiv(Cin, 7);
    if (hasDs) { int c2 = 64 * cdiv(Cin, 64); if (c2 > Cpad) Cpad = c2; }
    binarize(cur, Cin, Cpad, H);

    binconv(Cpad, p.w1, nxt, Cin, H, Cout, 3, st, 1);
    bnorm(nxt, p.b1g, p.b1b, Cout, Ho * Ho, 1);

    const float* res = cur;
    if (hasDs) {
      binconv(Cpad, p.dsw, F2, Cin, H, Cout, 1, st, 0);
      bnorm(F2, p.dsg, p.dsb, Cout, Ho * Ho, 0);
      res = F2;
    }

    int Cpad2 = 7 * cdiv(Cout, 7);
    binarize(nxt, Cout, Cpad2, Ho);
    binconv(Cpad2, p.w2, nxt, Cout, Ho, Cout, 3, 1, 1);

    long long tot = (long long)N * Cout * Ho * Ho;
    k_add<<<(unsigned)((tot + 255) / 256), 256, 0, stream>>>(nxt, res, tot);
    if (dbt) bnorm(nxt, p.b2g, p.b2b, Cout, Ho * Ho, 1);

    float* t = cur; cur = nxt; nxt = t;
  }

  k_avgpool<<<(unsigned)((N * 320 + 255) / 256), 256, 0, stream>>>(
      cur, POOL, N, 320, 64);
  bnorm(POOL, BN2G, BN2B, 320, 1, 1);
  k_fc<<<(unsigned)((N * 10 + 255) / 256), 256, 0, stream>>>(
      POOL, FCW, FCB, FCO, N, 320, 10);
  bnorm(FCO, BN3G, BN3B, 10, 1, 0);
  k_logsoftmax<<<1, 256, 0, stream>>>(FCO, (float*)d_out, N, 10);
}
